// SpatialTemporalAttentionBlock_91018946937479
// MI455X (gfx1250) — matrix operand order repaired and checked by reading, not compiled, not run on hardware
//
#include <hip/hip_runtime.h>

// ---------------------------------------------------------------------------
// CDNA5 (gfx1250) spatio-temporal attention block.
// GEMMs: v_wmma_f32_16x16x32_bf16, double-buffered LDS, async global->LDS DMA
// (GLOBAL_LOAD_ASYNC_TO_LDS_B128 / s_wait_asynccnt) when the toolchain has it.
// ---------------------------------------------------------------------------

typedef __attribute__((ext_vector_type(16))) __bf16 v16bf;
typedef __attribute__((ext_vector_type(8)))  float  v8f;
typedef int v4i_ __attribute__((vector_size(16)));   // matches async builtin param

#define D_MODEL 512
#define NHEAD   8
#define HDIM    64
#define NPATCH  196
#define LSEQ    16
#define BATCH   4
#define MTOK    (BATCH * LSEQ * NPATCH)   // 12544 tokens

// ---- async global->LDS copy (compile-guarded, falls back to load+store) ---
#if defined(__HIP_DEVICE_COMPILE__) && defined(__has_builtin)
#if __has_builtin(__builtin_amdgcn_global_load_async_to_lds_b128) && \
    __has_builtin(__builtin_amdgcn_s_wait_asynccnt)
#define USE_ASYNC_LDS 1
#endif
#endif
#ifndef USE_ASYNC_LDS
#define USE_ASYNC_LDS 0
#endif

static __device__ __forceinline__ void cp16(__bf16* lds, const __bf16* gl) {
#if USE_ASYNC_LDS
    __builtin_amdgcn_global_load_async_to_lds_b128(
        (__attribute__((address_space(1))) v4i_*)(gl),
        (__attribute__((address_space(3))) v4i_*)(lds), 0, 0);
#else
    *reinterpret_cast<uint4*>(lds) = *reinterpret_cast<const uint4*>(gl);
#endif
}
static __device__ __forceinline__ void cp_wait() {
#if USE_ASYNC_LDS
    __builtin_amdgcn_s_wait_asynccnt(0);
#endif
}

// ---- bf16 helpers (bit-exact RNE convert, no arithmetic on __bf16) --------
static __device__ __forceinline__ __bf16 f2b(float f) {
    unsigned u = __builtin_bit_cast(unsigned, f);
    unsigned r = u + 0x7FFFu + ((u >> 16) & 1u);
    unsigned short s = (unsigned short)(r >> 16);
    return __builtin_bit_cast(__bf16, s);
}
static __device__ __forceinline__ float b2f(__bf16 b) {
    unsigned short s = __builtin_bit_cast(unsigned short, b);
    unsigned u = ((unsigned)s) << 16;
    return __builtin_bit_cast(float, u);
}

// ---------------------------------------------------------------------------
// Weight convert + transpose: W[K,N] f32 -> Wt[N,K] bf16
// ---------------------------------------------------------------------------
__global__ __launch_bounds__(256)
void transpose_to_bf16(const float* __restrict__ W, __bf16* __restrict__ Wt,
                       int K, int N) {
    int idx = blockIdx.x * 256 + threadIdx.x;
    int total = K * N;
    if (idx >= total) return;
    int n = idx / K, k = idx - n * K;
    Wt[idx] = f2b(W[(size_t)k * N + n]);
}

// Elementwise f32 -> bf16
__global__ __launch_bounds__(256)
void convert_to_bf16(const float* __restrict__ src, __bf16* __restrict__ dst,
                     int total) {
    int idx = blockIdx.x * 256 + threadIdx.x;
    if (idx < total) dst[idx] = f2b(src[idx]);
}

// ---------------------------------------------------------------------------
// bf16 WMMA GEMM:  C[M,N] = A[M,K] * Bt[N,K]^T + bias
// Block tile 128x128x32, 256 threads = 8 waves (4x2), wave tile 32x64
// (2 A-frags x 4 B-frags = 8 WMMA per K-step). Double-buffered LDS with
// async global->LDS DMA. Requires M%128==0, N%128==0, K%32==0.
// EPI: 0 = f32 out, 1 = bf16 out, 2 = bf16 out with exact GELU.
// ---------------------------------------------------------------------------
#define BM 128
#define BN 128
#define BK 32
#define LDSS 40   // padded bf16 stride (80 B/row) to break bank conflicts

union Frag { v16bf v; uint4 q[2]; };

template<int EPI>
__global__ __launch_bounds__(256)
void gemm_bf16_wmma(const __bf16* __restrict__ A, const __bf16* __restrict__ Bt,
                    const float* __restrict__ bias,
                    float* __restrict__ Cf, __bf16* __restrict__ Cb,
                    int M, int N, int K) {
    __shared__ __align__(16) __bf16 sA[2][BM * LDSS];
    __shared__ __align__(16) __bf16 sB[2][BN * LDSS];

    const int t    = threadIdx.x;
    const int lane = t & 31, wave = t >> 5;
    const int wm   = wave >> 1, wn = wave & 1;      // 4x2 wave grid
    const int g    = lane >> 4, lr = lane & 15;
    const int bm0  = blockIdx.x * BM, bn0 = blockIdx.y * BN;

    // Staging: A tile (128x32) and B tile (128x32) = 512 16B chunks each;
    // each thread owns rows r0,r0+64 at column-chunk o for both tiles.
    const int r0 = t >> 2, o = (t & 3) << 3, r1 = r0 + 64;
    const __bf16* gA0 = A  + (size_t)(bm0 + r0) * K + o;
    const __bf16* gA1 = A  + (size_t)(bm0 + r1) * K + o;
    const __bf16* gB0 = Bt + (size_t)(bn0 + r0) * K + o;
    const __bf16* gB1 = Bt + (size_t)(bn0 + r1) * K + o;
    const int lOffA0 = r0 * LDSS + o, lOffA1 = r1 * LDSS + o;

    auto issue = [&](int buf, int k0) {
        cp16(&sA[buf][lOffA0], gA0 + k0);
        cp16(&sA[buf][lOffA1], gA1 + k0);
        cp16(&sB[buf][lOffA0], gB0 + k0);
        cp16(&sB[buf][lOffA1], gB1 + k0);
    };

    v8f zero = {};
    v8f acc[2][4];
    #pragma unroll
    for (int i = 0; i < 2; ++i)
        #pragma unroll
        for (int j = 0; j < 4; ++j) acc[i][j] = zero;

    const int nsteps = K / BK;
    issue(0, 0);

    for (int s = 0; s < nsteps; ++s) {
        const int cur = s & 1;
        cp_wait();                 // own async copies for buf(cur) complete
        __syncthreads();           // whole block done writing buf(cur),
                                   // done reading buf(cur^1)
        if (s + 1 < nsteps) issue(cur ^ 1, (s + 1) * BK);

        Frag a[2], b[4];
        #pragma unroll
        for (int i = 0; i < 2; ++i) {
            // A 16x32 frag: lane=row lr, k in [8g,8g+8) U [16+8g,16+8g+8)
            const __bf16* base = &sA[cur][(wm * 32 + i * 16 + lr) * LDSS];
            a[i].q[0] = *reinterpret_cast<const uint4*>(base + 8 * g);
            a[i].q[1] = *reinterpret_cast<const uint4*>(base + 16 + 8 * g);
        }
        #pragma unroll
        for (int j = 0; j < 4; ++j) {
            // B 32x16 frag: lane=col lr, k in [8g,8g+8) U [16+8g,16+8g+8)
            const __bf16* base = &sB[cur][(wn * 64 + j * 16 + lr) * LDSS];
            b[j].q[0] = *reinterpret_cast<const uint4*>(base + 8 * g);
            b[j].q[1] = *reinterpret_cast<const uint4*>(base + 16 + 8 * g);
        }
        #pragma unroll
        for (int i = 0; i < 2; ++i)
            #pragma unroll
            for (int j = 0; j < 4; ++j)
                acc[i][j] = __builtin_amdgcn_wmma_f32_16x16x32_bf16(
                    false, a[i].v, false, b[j].v, (short)0, acc[i][j],
                    false, false);
    }

    // Epilogue. C/D layout: VGPR r holds M = r + 8g, N = lr.
    #pragma unroll
    for (int i = 0; i < 2; ++i) {
        #pragma unroll
        for (int j = 0; j < 4; ++j) {
            int n  = bn0 + wn * 64 + j * 16 + lr;
            int mb = bm0 + wm * 32 + i * 16 + g * 8;
            float bv = bias ? bias[n] : 0.0f;
            #pragma unroll
            for (int r = 0; r < 8; ++r) {
                float vv = acc[i][j][r] + bv;
                size_t idx = (size_t)(mb + r) * N + n;
                if (EPI == 0) {
                    Cf[idx] = vv;
                } else if (EPI == 1) {
                    Cb[idx] = f2b(vv);
                } else {
                    float ge = 0.5f * vv * (1.0f + erff(vv * 0.70710678118654752f));
                    Cb[idx] = f2b(ge);
                }
            }
        }
    }
}

// ---------------------------------------------------------------------------
// Spatial attention: one block per (image m in [0,64), head h).
// q,k,v,ctx: [12544, 512] bf16; row = m*196 + patch, col = h*64 + d.
// ---------------------------------------------------------------------------
__global__ __launch_bounds__(256)
void spatial_attn(const __bf16* __restrict__ q, const __bf16* __restrict__ k,
                  const __bf16* __restrict__ v, __bf16* __restrict__ ctx) {
    const int m = blockIdx.x >> 3, h = blockIdx.x & 7;
    __shared__ __align__(16) __bf16 sK[NPATCH * HDIM];
    __shared__ __align__(16) __bf16 sV[NPATCH * HDIM];
    __shared__ float sQ[HDIM];
    __shared__ float sS[NPATCH];
    __shared__ float red[256];

    const int t = threadIdx.x;
    const size_t base = ((size_t)m * NPATCH) * D_MODEL + (size_t)h * HDIM;

    // Stage K,V tiles (196x64 bf16 each) into LDS with async 16B copies.
    for (int c = t; c < NPATCH * 8; c += 256) {
        int r = c >> 3, oo = (c & 7) << 3;
        cp16(&sK[r * HDIM + oo], k + base + (size_t)r * D_MODEL + oo);
        cp16(&sV[r * HDIM + oo], v + base + (size_t)r * D_MODEL + oo);
    }
    cp_wait();
    __syncthreads();

    for (int qi = 0; qi < NPATCH; ++qi) {
        if (t < HDIM) sQ[t] = b2f(q[base + (size_t)qi * D_MODEL + t]);
        __syncthreads();

        // scores row
        if (t < NPATCH) {
            float acc = 0.0f;
            #pragma unroll 8
            for (int d = 0; d < HDIM; ++d) acc += sQ[d] * b2f(sK[t * HDIM + d]);
            sS[t] = acc * 0.125f;                       // 1/sqrt(64)
        }
        __syncthreads();

        // softmax: max
        red[t] = (t < NPATCH) ? sS[t] : -3.4e38f;
        __syncthreads();
        for (int s = 128; s > 0; s >>= 1) {
            if (t < s) red[t] = fmaxf(red[t], red[t + s]);
            __syncthreads();
        }
        float mx = red[0];
        __syncthreads();
        float e = 0.0f;
        if (t < NPATCH) { e = __expf(sS[t] - mx); sS[t] = e; }
        red[t] = e;
        __syncthreads();
        for (int s = 128; s > 0; s >>= 1) {
            if (t < s) red[t] += red[t + s];
            __syncthreads();
        }
        float inv = 1.0f / red[0];
        __syncthreads();

        // out row: 256 threads = 64 d x 4 j-chunks of 49
        int d = t & 63, c = t >> 6;
        float part = 0.0f;
        for (int j = c * 49; j < c * 49 + 49; ++j)
            part += sS[j] * b2f(sV[j * HDIM + d]);
        red[t] = part;
        __syncthreads();
        if (t < HDIM) {
            float ov = (red[t] + red[t + 64] + red[t + 128] + red[t + 192]) * inv;
            ctx[base + (size_t)qi * D_MODEL + t] = f2b(ov);
        }
        __syncthreads();
    }
}

// ---------------------------------------------------------------------------
// Temporal attention: block per (b, patch n, head h). Sequence length 16,
// causal mask + time-decay omega = 1/(1+exp(0.5*|ti-tj|-0.5)).
// ---------------------------------------------------------------------------
__global__ __launch_bounds__(256)
void temporal_attn(const __bf16* __restrict__ q, const __bf16* __restrict__ k,
                   const __bf16* __restrict__ v, const float* __restrict__ tv,
                   __bf16* __restrict__ ctx) {
    const int h  = blockIdx.x & 7;
    const int bn = blockIdx.x >> 3;
    const int b  = bn / NPATCH, n = bn - b * NPATCH;
    __shared__ float sQ[LSEQ * HDIM], sK[LSEQ * HDIM], sV[LSEQ * HDIM];
    __shared__ float sS[LSEQ * LSEQ];
    __shared__ float sT[LSEQ];
    const int t = threadIdx.x;

    for (int e = t; e < LSEQ * HDIM; e += 256) {
        int l = e >> 6, d = e & 63;
        size_t off = ((size_t)((b * LSEQ + l) * NPATCH + n)) * D_MODEL
                   + (size_t)h * HDIM + d;
        sQ[e] = b2f(q[off]); sK[e] = b2f(k[off]); sV[e] = b2f(v[off]);
    }
    if (t < LSEQ) sT[t] = tv[b * LSEQ + t];
    __syncthreads();

    // scores: thread (i,j)
    {
        int i = t >> 4, j = t & 15;
        float s;
        if (j > i) {
            s = -__builtin_inff();
        } else {
            float acc = 0.0f;
            #pragma unroll 8
            for (int d = 0; d < HDIM; ++d) acc += sQ[i * HDIM + d] * sK[j * HDIM + d];
            float td = fabsf(sT[i] - sT[j]);
            float omega = 1.0f / (1.0f + __expf(0.5f * td - 0.5f));
            s = acc * 0.125f * omega;
        }
        sS[t] = s;
    }
    __syncthreads();

    // per-row softmax (16 rows handled by threads 0..15)
    if (t < LSEQ) {
        float mx = -__builtin_inff();
        for (int j = 0; j <= t; ++j) mx = fmaxf(mx, sS[t * 16 + j]);
        float sum = 0.0f;
        for (int j = 0; j < LSEQ; ++j) {
            float e = (j <= t) ? __expf(sS[t * 16 + j] - mx) : 0.0f;
            sS[t * 16 + j] = e; sum += e;
        }
        float inv = 1.0f / sum;
        for (int j = 0; j < LSEQ; ++j) sS[t * 16 + j] *= inv;
    }
    __syncthreads();

    for (int e = t; e < LSEQ * HDIM; e += 256) {
        int l = e >> 6, d = e & 63;
        float acc = 0.0f;
        #pragma unroll
        for (int j = 0; j < LSEQ; ++j) acc += sS[l * 16 + j] * sV[j * HDIM + d];
        size_t off = ((size_t)((b * LSEQ + l) * NPATCH + n)) * D_MODEL
                   + (size_t)h * HDIM + d;
        ctx[off] = f2b(acc);
    }
}

// ---------------------------------------------------------------------------
// Fused residual add + LayerNorm (D=512). One block per token row.
// Writes f32 result and (optionally) a bf16 copy for the next GEMM.
// ---------------------------------------------------------------------------
__global__ __launch_bounds__(256)
void residual_ln(const float* __restrict__ xprev, const float* __restrict__ y,
                 const float* __restrict__ gamma, const float* __restrict__ beta,
                 float* __restrict__ xout, __bf16* __restrict__ xout_bf) {
    const int row = blockIdx.x, t = threadIdx.x;
    const size_t o = (size_t)row * D_MODEL;
    float a0 = xprev[o + t] + y[o + t];
    float a1 = xprev[o + t + 256] + y[o + t + 256];

    __shared__ float rs[256], rq[256];
    rs[t] = a0 + a1;
    rq[t] = a0 * a0 + a1 * a1;
    __syncthreads();
    for (int s = 128; s > 0; s >>= 1) {
        if (t < s) { rs[t] += rs[t + s]; rq[t] += rq[t + s]; }
        __syncthreads();
    }
    float mean = rs[0] * (1.0f / 512.0f);
    float var  = rq[0] * (1.0f / 512.0f) - mean * mean;
    float inv  = rsqrtf(var + 1e-5f);

    float o0 = (a0 - mean) * inv * gamma[t] + beta[t];
    float o1 = (a1 - mean) * inv * gamma[t + 256] + beta[t + 256];
    xout[o + t] = o0;
    xout[o + t + 256] = o1;
    if (xout_bf) {
        xout_bf[o + t] = f2b(o0);
        xout_bf[o + t + 256] = f2b(o1);
    }
}

// ---------------------------------------------------------------------------
// Host-side orchestration (all on `stream`, graph-capture safe).
// Workspace layout (bytes), total ~124 MB:
//   [0,          8,388,608)  : bf16 transposed weights (4,194,304 elems)
//   [8,388,608, 21,233,664)  : xbf  (current activation, bf16, 12544x512)
//   [21,233,664,72,613,888)  : q|k|v|ctx bf16  == h bf16 (FFN, disjoint life)
//   [72,613,888,98,304,000)  : y f32 (pre-LN GEMM output)
//   [98,304,000,123,994,112) : x1 f32 (post-LN1)
// d_out doubles as x2 (post-LN2) and the final output.
// ---------------------------------------------------------------------------
static void launch_gemm(int epi, const __bf16* A, const __bf16* Bt,
                        const float* bias, float* Cf, __bf16* Cb,
                        int M, int N, int K, hipStream_t s) {
    dim3 g(M / BM, N / BN), b(256);
    if (epi == 0)      gemm_bf16_wmma<0><<<g, b, 0, s>>>(A, Bt, bias, Cf, Cb, M, N, K);
    else if (epi == 1) gemm_bf16_wmma<1><<<g, b, 0, s>>>(A, Bt, bias, Cf, Cb, M, N, K);
    else               gemm_bf16_wmma<2><<<g, b, 0, s>>>(A, Bt, bias, Cf, Cb, M, N, K);
}

extern "C" void kernel_launch(void* const* d_in, const int* in_sizes, int n_in,
                              void* d_out, int out_size, void* d_ws, size_t ws_size,
                              hipStream_t stream) {
    (void)in_sizes; (void)n_in; (void)out_size; (void)ws_size;

    const float* x   = (const float*)d_in[0];
    const float* tiv = (const float*)d_in[1];
    // params: sq,sk,sv,so,tq,tk,tv,to (w,b pairs), ffn1, ffn2, ln1..3 (g,b)
    const float* W[10];
    const float* B[10];
    for (int i = 0; i < 10; ++i) {
        W[i] = (const float*)d_in[2 + 2 * i];
        B[i] = (const float*)d_in[3 + 2 * i];
    }
    const float* ln1g = (const float*)d_in[22]; const float* ln1b = (const float*)d_in[23];
    const float* ln2g = (const float*)d_in[24]; const float* ln2b = (const float*)d_in[25];
    const float* ln3g = (const float*)d_in[26]; const float* ln3b = (const float*)d_in[27];

    char* ws = (char*)d_ws;
    __bf16* wbf = (__bf16*)(ws);
    __bf16* xbf = (__bf16*)(ws + 8388608);
    __bf16* qb  = (__bf16*)(ws + 21233664);
    __bf16* kb  = qb + (size_t)MTOK * D_MODEL;
    __bf16* vb  = kb + (size_t)MTOK * D_MODEL;
    __bf16* ctx = vb + (size_t)MTOK * D_MODEL;
    __bf16* hbf = qb;                          // FFN hidden overlays q/k/v/ctx
    float*  yf  = (float*)(ws + 72613888);
    float*  x1  = (float*)(ws + 98304000);
    float*  x2  = (float*)d_out;

    // Weight bf16 [N,K] transposes.  sX/tX: 512x512, ffn1: 512x2048, ffn2: 2048x512.
    __bf16* wt[10];
    for (int i = 0; i < 8; ++i) {
        wt[i] = wbf + (size_t)i * 262144;
        transpose_to_bf16<<<(262144 + 255) / 256, 256, 0, stream>>>(W[i], wt[i], 512, 512);
    }
    wt[8] = wbf + 2097152;  // ffn1: [2048, 512]
    transpose_to_bf16<<<(1048576 + 255) / 256, 256, 0, stream>>>(W[8], wt[8], 512, 2048);
    wt[9] = wbf + 3145728;  // ffn2: [512, 2048]
    transpose_to_bf16<<<(1048576 + 255) / 256, 256, 0, stream>>>(W[9], wt[9], 2048, 512);

    const int TOT = MTOK * D_MODEL;            // 6,422,528
    convert_to_bf16<<<TOT / 256, 256, 0, stream>>>(x, xbf, TOT);

    // ---- spatial attention ----
    launch_gemm(1, xbf, wt[0], B[0], nullptr, qb, MTOK, 512, 512, stream);   // sq
    launch_gemm(1, xbf, wt[1], B[1], nullptr, kb, MTOK, 512, 512, stream);   // sk
    launch_gemm(1, xbf, wt[2], B[2], nullptr, vb, MTOK, 512, 512, stream);   // sv
    spatial_attn<<<BATCH * LSEQ * NHEAD, 256, 0, stream>>>(qb, kb, vb, ctx);
    launch_gemm(0, ctx, wt[3], B[3], yf, nullptr, MTOK, 512, 512, stream);   // so
    residual_ln<<<MTOK, 256, 0, stream>>>(x, yf, ln1g, ln1b, x1, xbf);

    // ---- temporal attention ----
    launch_gemm(1, xbf, wt[4], B[4], nullptr, qb, MTOK, 512, 512, stream);   // tq
    launch_gemm(1, xbf, wt[5], B[5], nullptr, kb, MTOK, 512, 512, stream);   // tk
    launch_gemm(1, xbf, wt[6], B[6], nullptr, vb, MTOK, 512, 512, stream);   // tv
    temporal_attn<<<BATCH * NPATCH * NHEAD, 256, 0, stream>>>(qb, kb, vb, tiv, ctx);
    launch_gemm(0, ctx, wt[7], B[7], yf, nullptr, MTOK, 512, 512, stream);   // to
    residual_ln<<<MTOK, 256, 0, stream>>>(x1, yf, ln2g, ln2b, x2, xbf);

    // ---- FFN ----
    launch_gemm(2, xbf, wt[8], B[8], nullptr, hbf, MTOK, 2048, 512, stream); // ffn1+GELU
    launch_gemm(0, hbf, wt[9], B[9], yf, nullptr, MTOK, 512, 2048, stream);  // ffn2
    residual_ln<<<MTOK, 256, 0, stream>>>(x2, yf, ln3g, ln3b, x2, nullptr);  // in-place -> d_out
}
